// NaiveSwitchTransformerEncoderLayer_80814104642412
// MI455X (gfx1250) — matrix operand order repaired and checked by reading, not compiled, not run on hardware
//
#include <hip/hip_runtime.h>
#include <hip/hip_bf16.h>

// Switch-Transformer encoder layer for MI455X (gfx1250, wave32, WMMA).
// Dense math via v_wmma_f32_16x16x32_bf16; bf16 operands pre-converted; GEMM
// inner loop: double-buffered LDS, 4 WMMA per wave per K-step, 1 barrier/step.

typedef __attribute__((ext_vector_type(16))) __bf16 v16bf;
typedef __attribute__((ext_vector_type(8)))  float  v8f;
typedef __attribute__((ext_vector_type(8)))  __bf16 bf16x8;
typedef __attribute__((ext_vector_type(4)))  __bf16 bf16x4;
typedef __attribute__((ext_vector_type(4)))  float  f32x4;

#define D_MODEL 1024
#define NHEAD   4
#define HDIM    256
#define DFF     4096
#define NEXP    8
#define SEQ     512
#define BATCH   8
#define NTOK    (SEQ * BATCH)       // 4096 tokens

#define LDS_PAD 40                  // bf16 per LDS row: 80B rows, 16B aligned, 20-dword stride

__device__ __forceinline__ __bf16 f2bf(float f) {
  unsigned u; __builtin_memcpy(&u, &f, 4);
  u += 0x7FFFu + ((u >> 16) & 1u);          // round-to-nearest-even
  unsigned short h = (unsigned short)(u >> 16);
  __bf16 r; __builtin_memcpy(&r, &h, 2);
  return r;
}

union FragU { v16bf v; uint4 q[2]; };

#define WMMA_BF16(A_, B_, C_) \
  __builtin_amdgcn_wmma_f32_16x16x32_bf16(false, (A_), false, (B_), (short)0, (C_), false, false)

// Load a 16x32 A-fragment / 32x16 B-fragment from an LDS tile (2x ds_load_b128).
__device__ __forceinline__ void load_afrag(FragU& f, const unsigned* base, int mrow, int khalf) {
  const unsigned* p = base + mrow * (LDS_PAD / 2);
  f.q[0] = *(const uint4*)(p + 4 * khalf);
  f.q[1] = *(const uint4*)(p + 8 + 4 * khalf);
}
__device__ __forceinline__ void load_bfrag(FragU& f, const unsigned* base, int nloc, int khalf) {
  const unsigned* p = base + nloc * (LDS_PAD / 2);
  f.q[0] = *(const uint4*)(p + 4 * khalf);
  f.q[1] = *(const uint4*)(p + 8 + 4 * khalf);
}

// ---------------------------------------------------------------------------
// bf16 WMMA GEMM:  C = alpha*(A@B) [+bias] [+R]
//   A: bf16 (M,K) row-major, stride lda
//   B: NT_B=false -> bf16 (N,K) row-major ("W^T"), stride ldb
//      NT_B=true  -> bf16 (K,N) row-major, stride ldb
// 256 threads = 8 waves; block tile 128x64, K step 32, double-buffered LDS.
// Wave = 2x2 16x16 tiles (2 A-frags x 2 B-frags -> 4 WMMA / step).
// Batched over blockIdx.z = zb*ZH + zh.
// ---------------------------------------------------------------------------
template <bool NT_B, bool BIAS, bool RES, bool OUT_BF16>
__global__ __launch_bounds__(256)
void wmma_gemm(const __bf16* __restrict__ A, const __bf16* __restrict__ Bm,
               const float* __restrict__ bias, const float* __restrict__ R,
               void* __restrict__ Cout,
               int M, int N, int K, int lda, int ldb, int ldc, int ldr,
               float alpha, int ZH,
               long long sAb, long long sAh, long long sBb, long long sBh,
               long long sCb, long long sCh) {
  const int zb = blockIdx.z / ZH, zh = blockIdx.z % ZH;
  A  += zb * sAb + zh * sAh;
  Bm += zb * sBb + zh * sBh;
  const long long coff = zb * sCb + zh * sCh;

  const int tile_m = blockIdx.y * 128;
  const int tile_n = blockIdx.x * 64;

  __shared__ __bf16 As[2][128 * LDS_PAD];
  __shared__ __bf16 Bs[2][64 * LDS_PAD];

  const int wave = threadIdx.x >> 5;
  const int lane = threadIdx.x & 31;

  // staging coords: A tile 128x32 (2 rows/thread), B tile 64x32 (1 row/thread)
  const int ar0 = threadIdx.x >> 2;              // 0..63
  const int ar1 = ar0 + 64;
  const int ac  = (threadIdx.x & 3) * 8;         // 0,8,16,24
  const bool ok0 = (tile_m + ar0) < M;
  const bool ok1 = (tile_m + ar1) < M;
  const __bf16* Arow0 = A + (long long)(tile_m + ar0) * lda + ac;
  const __bf16* Arow1 = A + (long long)(tile_m + ar1) * lda + ac;
  const int kr = threadIdx.x >> 3;               // NT: k row 0..31
  const int nc = (threadIdx.x & 7) * 8;          // NT: n offset
  const __bf16* Brow = NT_B ? (Bm + (long long)kr * ldb + tile_n + nc)
                            : (Bm + (long long)(tile_n + ar0) * ldb + ac);

  // ---- prologue: stage K-tile 0 into buffer 0 ----
  {
    bf16x8 v0 = {}, v1 = {};
    if (ok0) v0 = *(const bf16x8*)(Arow0);
    if (ok1) v1 = *(const bf16x8*)(Arow1);
    *(bf16x8*)(&As[0][ar0 * LDS_PAD + ac]) = v0;
    *(bf16x8*)(&As[0][ar1 * LDS_PAD + ac]) = v1;
    bf16x8 vb = *(const bf16x8*)(Brow);
    if (!NT_B) {
      *(bf16x8*)(&Bs[0][ar0 * LDS_PAD + ac]) = vb;
    } else {
#pragma unroll
      for (int j = 0; j < 8; ++j) Bs[0][(nc + j) * LDS_PAD + kr] = vb[j];
    }
  }
  __syncthreads();

  v8f acc00 = {}, acc01 = {}, acc10 = {}, acc11 = {};
  const int mrow0 = (wave & 3) * 32 + (lane & 15);
  const int mrow1 = mrow0 + 16;
  const int khalf = lane >> 4;
  const int nloc0 = (wave >> 2) * 32 + (lane & 15);
  const int nloc1 = nloc0 + 16;

  for (int k0 = 0; k0 < K; k0 += 32) {
    const int cur = (k0 >> 5) & 1, nxt = cur ^ 1;
    const bool more = (k0 + 32) < K;

    // fetch next K-tile into registers (overlaps with WMMA below)
    bf16x8 va0 = {}, va1 = {}, vb = {};
    if (more) {
      const int kn = k0 + 32;
      if (ok0) { va0 = *(const bf16x8*)(Arow0 + kn); __builtin_prefetch(Arow0 + kn + 32, 0, 1); }
      if (ok1)   va1 = *(const bf16x8*)(Arow1 + kn);
      if (!NT_B) { vb = *(const bf16x8*)(Brow + kn); __builtin_prefetch(Brow + kn + 32, 0, 1); }
      else         vb = *(const bf16x8*)(Brow + (long long)kn * ldb);
    }

    // fragments from current buffer (2x ds_load_b128 each) + 4 WMMA
    const unsigned* Au = (const unsigned*)As[cur];
    const unsigned* Bu = (const unsigned*)Bs[cur];
    FragU a0, a1, b0, b1;
    load_afrag(a0, Au, mrow0, khalf);
    load_afrag(a1, Au, mrow1, khalf);
    load_bfrag(b0, Bu, nloc0, khalf);
    load_bfrag(b1, Bu, nloc1, khalf);
    acc00 = WMMA_BF16(a0.v, b0.v, acc00);
    acc01 = WMMA_BF16(a0.v, b1.v, acc01);
    acc10 = WMMA_BF16(a1.v, b0.v, acc10);
    acc11 = WMMA_BF16(a1.v, b1.v, acc11);

    if (more) {
      *(bf16x8*)(&As[nxt][ar0 * LDS_PAD + ac]) = va0;
      *(bf16x8*)(&As[nxt][ar1 * LDS_PAD + ac]) = va1;
      if (!NT_B) {
        *(bf16x8*)(&Bs[nxt][ar0 * LDS_PAD + ac]) = vb;
      } else {
#pragma unroll
        for (int j = 0; j < 8; ++j) Bs[nxt][(nc + j) * LDS_PAD + kr] = vb[j];
      }
    }
    __syncthreads();     // single barrier per K-step (double buffered)
  }

  // ---- epilogue ----
  const int col0  = tile_n + nloc0;
  const int col1  = tile_n + nloc1;
  const int rbase = (lane >> 4) * 8;
#pragma unroll
  for (int half = 0; half < 2; ++half) {
    const v8f& c0 = half ? acc10 : acc00;
    const v8f& c1 = half ? acc11 : acc01;
    const int rowb = tile_m + (wave & 3) * 32 + half * 16 + rbase;
#pragma unroll
    for (int r = 0; r < 8; ++r) {
      int row = rowb + r;
      if (row < M) {
        float v0 = c0[r] * alpha, v1 = c1[r] * alpha;
        if (BIAS) { v0 += bias[col0]; v1 += bias[col1]; }
        if (RES) {
          v0 += R[coff + (long long)row * ldr + col0];
          v1 += R[coff + (long long)row * ldr + col1];
        }
        if (OUT_BF16) {
          __bf16* C = (__bf16*)Cout;
          C[coff + (long long)row * ldc + col0] = f2bf(v0);
          C[coff + (long long)row * ldc + col1] = f2bf(v1);
        } else {
          float* C = (float*)Cout;
          C[coff + (long long)row * ldc + col0] = v0;
          C[coff + (long long)row * ldc + col1] = v1;
        }
      }
    }
  }
}

// ---------------------------------------------------------------------------
// MoE grouped GEMM: rows i in [ebase[e], ebase[e+1]); A row = GATHER?perm[i]:i,
// C row = SCATTER?perm[i]:i.  B = pre-transposed expert weight (N,K) bf16.
// Same 128x64 double-buffered scheme.
// ---------------------------------------------------------------------------
template <bool GATHER_A, bool SCATTER_C, bool RELU, bool OUT_BF16>
__global__ __launch_bounds__(256)
void moe_gemm(const __bf16* __restrict__ Asrc, const __bf16* __restrict__ WT,
              const float* __restrict__ bias, void* __restrict__ Cout,
              const int* __restrict__ perm, const int* __restrict__ ebase,
              int N, int K, int lda, int ldc, long long Wstride) {
  const int e  = blockIdx.z;
  const int lo = ebase[e], hi = ebase[e + 1];
  const int row0 = lo + blockIdx.y * 128;
  if (row0 >= hi) return;                        // empty tile: one scalar load

  const __bf16* We = WT + (long long)e * Wstride;
  const float*  be = bias + (long long)e * N;
  const int tile_n = blockIdx.x * 64;

  __shared__ __bf16 As[2][128 * LDS_PAD];
  __shared__ __bf16 Bs[2][64 * LDS_PAD];

  const int wave = threadIdx.x >> 5;
  const int lane = threadIdx.x & 31;

  const int ar0 = threadIdx.x >> 2;
  const int ar1 = ar0 + 64;
  const int ac  = (threadIdx.x & 3) * 8;
  const int gi0 = row0 + ar0, gi1 = row0 + ar1;
  const bool ok0 = gi0 < hi, ok1 = gi1 < hi;
  const int sr0 = ok0 ? (GATHER_A ? perm[gi0] : gi0) : 0;
  const int sr1 = ok1 ? (GATHER_A ? perm[gi1] : gi1) : 0;
  const __bf16* Arow0 = Asrc + (long long)sr0 * lda + ac;
  const __bf16* Arow1 = Asrc + (long long)sr1 * lda + ac;
  const __bf16* Brow  = We + (long long)(tile_n + ar0) * K + ac;

  {
    bf16x8 v0 = {}, v1 = {};
    if (ok0) v0 = *(const bf16x8*)(Arow0);
    if (ok1) v1 = *(const bf16x8*)(Arow1);
    *(bf16x8*)(&As[0][ar0 * LDS_PAD + ac]) = v0;
    *(bf16x8*)(&As[0][ar1 * LDS_PAD + ac]) = v1;
    *(bf16x8*)(&Bs[0][ar0 * LDS_PAD + ac]) = *(const bf16x8*)(Brow);
  }
  __syncthreads();

  v8f acc00 = {}, acc01 = {}, acc10 = {}, acc11 = {};
  const int mrow0 = (wave & 3) * 32 + (lane & 15);
  const int mrow1 = mrow0 + 16;
  const int khalf = lane >> 4;
  const int nloc0 = (wave >> 2) * 32 + (lane & 15);
  const int nloc1 = nloc0 + 16;

  for (int k0 = 0; k0 < K; k0 += 32) {
    const int cur = (k0 >> 5) & 1, nxt = cur ^ 1;
    const bool more = (k0 + 32) < K;

    bf16x8 va0 = {}, va1 = {}, vb = {};
    if (more) {
      const int kn = k0 + 32;
      if (ok0) { va0 = *(const bf16x8*)(Arow0 + kn); __builtin_prefetch(Arow0 + kn + 32, 0, 1); }
      if (ok1)   va1 = *(const bf16x8*)(Arow1 + kn);
      vb = *(const bf16x8*)(Brow + kn);
      __builtin_prefetch(Brow + kn + 32, 0, 1);
    }

    const unsigned* Au = (const unsigned*)As[cur];
    const unsigned* Bu = (const unsigned*)Bs[cur];
    FragU a0, a1, b0, b1;
    load_afrag(a0, Au, mrow0, khalf);
    load_afrag(a1, Au, mrow1, khalf);
    load_bfrag(b0, Bu, nloc0, khalf);
    load_bfrag(b1, Bu, nloc1, khalf);
    acc00 = WMMA_BF16(a0.v, b0.v, acc00);
    acc01 = WMMA_BF16(a0.v, b1.v, acc01);
    acc10 = WMMA_BF16(a1.v, b0.v, acc10);
    acc11 = WMMA_BF16(a1.v, b1.v, acc11);

    if (more) {
      *(bf16x8*)(&As[nxt][ar0 * LDS_PAD + ac]) = va0;
      *(bf16x8*)(&As[nxt][ar1 * LDS_PAD + ac]) = va1;
      *(bf16x8*)(&Bs[nxt][ar0 * LDS_PAD + ac]) = vb;
    }
    __syncthreads();
  }

  const int col0  = tile_n + nloc0;
  const int col1  = tile_n + nloc1;
  const int rbase = (lane >> 4) * 8;
#pragma unroll
  for (int half = 0; half < 2; ++half) {
    const v8f& c0 = half ? acc10 : acc00;
    const v8f& c1 = half ? acc11 : acc01;
    const int gb = row0 + (wave & 3) * 32 + half * 16 + rbase;
#pragma unroll
    for (int r = 0; r < 8; ++r) {
      int g = gb + r;
      if (g < hi) {
        float v0 = c0[r] + be[col0];
        float v1 = c1[r] + be[col1];
        if (RELU) { v0 = fmaxf(v0, 0.f); v1 = fmaxf(v1, 0.f); }
        int crow = SCATTER_C ? perm[g] : g;
        if (OUT_BF16) {
          __bf16* C = (__bf16*)Cout;
          C[(long long)crow * ldc + col0] = f2bf(v0);
          C[(long long)crow * ldc + col1] = f2bf(v1);
        } else {
          float* C = (float*)Cout;
          C[(long long)crow * ldc + col0] = v0;
          C[(long long)crow * ldc + col1] = v1;
        }
      }
    }
  }
}

// ---------------------------------------------------------------------------
// fp32 -> bf16 elementwise (n4 = n/4)
// ---------------------------------------------------------------------------
__global__ __launch_bounds__(256)
void cvt_bf16(const float* __restrict__ in, __bf16* __restrict__ outp, long long n4) {
  long long i = (long long)blockIdx.x * 256 + threadIdx.x;
  if (i >= n4) return;
  f32x4 v = ((const f32x4*)in)[i];
  bf16x4 o;
#pragma unroll
  for (int j = 0; j < 4; ++j) o[j] = f2bf(v[j]);
  ((bf16x4*)outp)[i] = o;
}

// fp32 (K,N) -> bf16 (N,K) transpose-convert, batched over blockIdx.z
__global__ __launch_bounds__(256)
void cvt_bf16_t(const float* __restrict__ in, __bf16* __restrict__ outp,
                int K, int N, long long sIn, long long sOut) {
  __shared__ float t[32][33];
  const float* src = in + (long long)blockIdx.z * sIn;
  __bf16* dst = outp + (long long)blockIdx.z * sOut;
  const int k0 = blockIdx.y * 32, n0 = blockIdx.x * 32;
  for (int i = threadIdx.x; i < 1024; i += 256) {
    int r = i >> 5, c = i & 31;
    t[r][c] = src[(long long)(k0 + r) * N + n0 + c];
  }
  __syncthreads();
  for (int i = threadIdx.x; i < 1024; i += 256) {
    int r = i >> 5, c = i & 31;
    dst[(long long)(n0 + r) * K + k0 + c] = f2bf(t[c][r]);
  }
}

// ---------------------------------------------------------------------------
// Softmax over rows of 512 (fp32 in, bf16 probs out).
// ---------------------------------------------------------------------------
__global__ __launch_bounds__(128)
void softmax512(const float* __restrict__ s, __bf16* __restrict__ pout) {
  __shared__ float red[128];
  const float* p = s + (long long)blockIdx.x * SEQ;
  __bf16* q = pout + (long long)blockIdx.x * SEQ;
  float vals[4];
  float m = -1e30f;
#pragma unroll
  for (int j = 0; j < 4; ++j) m = fmaxf(m, p[threadIdx.x + j * 128]);
  red[threadIdx.x] = m; __syncthreads();
  for (int off = 64; off > 0; off >>= 1) {
    if (threadIdx.x < off) red[threadIdx.x] = fmaxf(red[threadIdx.x], red[threadIdx.x + off]);
    __syncthreads();
  }
  m = red[0]; __syncthreads();
  float sum = 0.f;
#pragma unroll
  for (int j = 0; j < 4; ++j) {
    float v = __expf(p[threadIdx.x + j * 128] - m);
    vals[j] = v; sum += v;
  }
  red[threadIdx.x] = sum; __syncthreads();
  for (int off = 64; off > 0; off >>= 1) {
    if (threadIdx.x < off) red[threadIdx.x] += red[threadIdx.x + off];
    __syncthreads();
  }
  float inv = 1.f / red[0];
#pragma unroll
  for (int j = 0; j < 4; ++j) q[threadIdx.x + j * 128] = f2bf(vals[j] * inv);
}

// ---------------------------------------------------------------------------
// LayerNorm: out = LN(X [+ Rs])*g + be ; optional bf16 mirror.
// ---------------------------------------------------------------------------
template <bool HASRES, bool WRITE_BF>
__global__ __launch_bounds__(256)
void layernorm_kernel(const float* __restrict__ X, const float* __restrict__ Rs,
                      const float* __restrict__ g, const float* __restrict__ be,
                      float* __restrict__ out, __bf16* __restrict__ out_bf) {
  __shared__ float red[256];
  const long long base = (long long)blockIdx.x * D_MODEL;
  float v[4]; float s = 0.f;
#pragma unroll
  for (int j = 0; j < 4; ++j) {
    int i = threadIdx.x + j * 256;
    float t = X[base + i];
    if (HASRES) t += Rs[base + i];
    v[j] = t; s += t;
  }
  red[threadIdx.x] = s; __syncthreads();
  for (int off = 128; off > 0; off >>= 1) {
    if (threadIdx.x < off) red[threadIdx.x] += red[threadIdx.x + off];
    __syncthreads();
  }
  float mu = red[0] * (1.f / D_MODEL); __syncthreads();
  float var = 0.f;
#pragma unroll
  for (int j = 0; j < 4; ++j) { float d = v[j] - mu; var += d * d; }
  red[threadIdx.x] = var; __syncthreads();
  for (int off = 128; off > 0; off >>= 1) {
    if (threadIdx.x < off) red[threadIdx.x] += red[threadIdx.x + off];
    __syncthreads();
  }
  float inv = rsqrtf(red[0] * (1.f / D_MODEL) + 1e-5f);
#pragma unroll
  for (int j = 0; j < 4; ++j) {
    int i = threadIdx.x + j * 256;
    float o = (v[j] - mu) * inv * g[i] + be[i];
    out[base + i] = o;
    if (WRITE_BF) out_bf[base + i] = f2bf(o);
  }
}

// ---------------------------------------------------------------------------
// MoE routing
// ---------------------------------------------------------------------------
__global__ void moe_init(int* counts) {
  if (threadIdx.x < NEXP && blockIdx.x == 0) counts[threadIdx.x] = 0;
}

__global__ __launch_bounds__(128)
void gate_kernel(const float* __restrict__ y, const float* __restrict__ Wg,
                 int* __restrict__ expert_id, int* __restrict__ counts) {
  int t = blockIdx.x * blockDim.x + threadIdx.x;
  if (t >= NTOK) return;
  const float* yr = y + (long long)t * D_MODEL;
  float best = -1e30f; int bi = 0;
  for (int e = 0; e < NEXP; ++e) {
    const float* w = Wg + (long long)e * D_MODEL;
    float s = 0.f;
    for (int d = 0; d < D_MODEL; ++d) s += yr[d] * w[d];
    if (s > best) { best = s; bi = e; }
  }
  expert_id[t] = bi;
  atomicAdd(&counts[bi], 1);
}

__global__ void moe_prefix(const int* __restrict__ counts, int* __restrict__ ebase,
                           int* __restrict__ cursor) {
  if (threadIdx.x == 0 && blockIdx.x == 0) {
    int a = 0;
    for (int e = 0; e < NEXP; ++e) { ebase[e] = a; cursor[e] = a; a += counts[e]; }
    ebase[NEXP] = a;
  }
}

__global__ __launch_bounds__(128)
void moe_scatter(const int* __restrict__ expert_id, int* __restrict__ cursor,
                 int* __restrict__ perm) {
  int t = blockIdx.x * blockDim.x + threadIdx.x;
  if (t >= NTOK) return;
  int e = expert_id[t];
  int pos = atomicAdd(&cursor[e], 1);
  perm[pos] = t;
}

// ---------------------------------------------------------------------------
extern "C" void kernel_launch(void* const* d_in, const int* in_sizes, int n_in,
                              void* d_out, int out_size, void* d_ws, size_t ws_size,
                              hipStream_t stream) {
  (void)in_sizes; (void)n_in; (void)out_size; (void)ws_size;
  const float* x    = (const float*)d_in[0];
  const float* Wqkv = (const float*)d_in[1];
  const float* bqkv = (const float*)d_in[2];
  const float* Wo   = (const float*)d_in[3];
  const float* bo   = (const float*)d_in[4];
  const float* g1   = (const float*)d_in[5];
  const float* be1  = (const float*)d_in[6];
  const float* g2   = (const float*)d_in[7];
  const float* be2  = (const float*)d_in[8];
  const float* Wg   = (const float*)d_in[9];
  const float* W1   = (const float*)d_in[10];
  const float* b1   = (const float*)d_in[11];
  const float* W2   = (const float*)d_in[12];
  const float* b2   = (const float*)d_in[13];
  float* out = (float*)d_out;

  // ---- workspace carve (all offsets 256B aligned) ----
  char* p = (char*)d_ws;
  auto carve = [&](size_t bytes) { char* r = p; p += (bytes + 255) & ~(size_t)255; return r; };
  __bf16* x_bf    = (__bf16*)carve((size_t)NTOK * D_MODEL * 2);
  __bf16* Wqkv_bf = (__bf16*)carve((size_t)3 * D_MODEL * D_MODEL * 2);
  __bf16* Wo_bf   = (__bf16*)carve((size_t)D_MODEL * D_MODEL * 2);
  __bf16* W1T     = (__bf16*)carve((size_t)NEXP * DFF * D_MODEL * 2);
  __bf16* W2T     = (__bf16*)carve((size_t)NEXP * D_MODEL * DFF * 2);
  __bf16* qkv_bf  = (__bf16*)carve((size_t)NTOK * 3 * D_MODEL * 2);
  float*  scores  = (float*)carve((size_t)BATCH * NHEAD * SEQ * SEQ * 4);
  __bf16* probs   = (__bf16*)carve((size_t)BATCH * NHEAD * SEQ * SEQ * 2);
  __bf16* o_bf    = (__bf16*)carve((size_t)NTOK * D_MODEL * 2);
  float*  y0      = (float*)carve((size_t)NTOK * D_MODEL * 4);
  float*  y1      = (float*)carve((size_t)NTOK * D_MODEL * 4);
  __bf16* y1_bf   = (__bf16*)carve((size_t)NTOK * D_MODEL * 2);
  __bf16* h_bf    = (__bf16*)carve((size_t)NTOK * DFF * 2);
  float*  moe     = (float*)carve((size_t)NTOK * D_MODEL * 4);
  int* expert_id  = (int*)carve(NTOK * 4);
  int* counts     = (int*)carve(NEXP * 4);
  int* ebase      = (int*)carve((NEXP + 1) * 4);
  int* cursor     = (int*)carve(NEXP * 4);
  int* perm       = (int*)carve(NTOK * 4);

  const int ldq = BATCH * 3 * D_MODEL;    // qkv token-row stride seen per (s)

  // ---- 0) one-pass bf16 conversion of inputs/weights ----
  cvt_bf16<<<(NTOK * D_MODEL / 4 + 255) / 256, 256, 0, stream>>>(x, x_bf, NTOK * D_MODEL / 4);
  cvt_bf16<<<(3 * D_MODEL * D_MODEL / 4 + 255) / 256, 256, 0, stream>>>(Wqkv, Wqkv_bf, 3 * D_MODEL * D_MODEL / 4);
  cvt_bf16<<<(D_MODEL * D_MODEL / 4 + 255) / 256, 256, 0, stream>>>(Wo, Wo_bf, D_MODEL * D_MODEL / 4);
  cvt_bf16_t<<<dim3(DFF / 32, D_MODEL / 32, NEXP), 256, 0, stream>>>(
      W1, W1T, D_MODEL, DFF, (long long)D_MODEL * DFF, (long long)DFF * D_MODEL);
  cvt_bf16_t<<<dim3(D_MODEL / 32, DFF / 32, NEXP), 256, 0, stream>>>(
      W2, W2T, DFF, D_MODEL, (long long)DFF * D_MODEL, (long long)D_MODEL * DFF);

  // ---- 1) QKV: qkv_bf = bf16(x @ Wqkv^T + bqkv) ----
  wmma_gemm<false, true, false, true><<<dim3(3 * D_MODEL / 64, NTOK / 128, 1), 256, 0, stream>>>(
      x_bf, Wqkv_bf, bqkv, nullptr, qkv_bf,
      NTOK, 3 * D_MODEL, D_MODEL, D_MODEL, D_MODEL, 3 * D_MODEL, 0,
      1.f, 1, 0, 0, 0, 0, 0, 0);

  // ---- 2) scores(b,h) = Q @ K^T / 16 ----
  wmma_gemm<false, false, false, false><<<dim3(SEQ / 64, SEQ / 128, BATCH * NHEAD), 256, 0, stream>>>(
      qkv_bf, qkv_bf + D_MODEL, nullptr, nullptr, scores,
      SEQ, SEQ, HDIM, ldq, ldq, SEQ, 0,
      0.0625f, NHEAD,
      3 * D_MODEL, HDIM, 3 * D_MODEL, HDIM,
      (long long)NHEAD * SEQ * SEQ, (long long)SEQ * SEQ);

  // ---- 3) softmax -> bf16 probs ----
  softmax512<<<BATCH * NHEAD * SEQ, 128, 0, stream>>>(scores, probs);

  // ---- 4) O = P @ V (V is (K=s, N=d) row-major -> NT path) ----
  wmma_gemm<true, false, false, true><<<dim3(HDIM / 64, SEQ / 128, BATCH * NHEAD), 256, 0, stream>>>(
      probs, qkv_bf + 2 * D_MODEL, nullptr, nullptr, o_bf,
      SEQ, HDIM, SEQ, SEQ, ldq, BATCH * D_MODEL, 0,
      1.f, NHEAD,
      (long long)NHEAD * SEQ * SEQ, (long long)SEQ * SEQ,
      3 * D_MODEL, HDIM, D_MODEL, HDIM);

  // ---- 5) out-proj + residual: y0 = o @ Wo^T + bo + x ----
  wmma_gemm<false, true, true, false><<<dim3(D_MODEL / 64, NTOK / 128, 1), 256, 0, stream>>>(
      o_bf, Wo_bf, bo, x, y0,
      NTOK, D_MODEL, D_MODEL, D_MODEL, D_MODEL, D_MODEL, D_MODEL,
      1.f, 1, 0, 0, 0, 0, 0, 0);

  // ---- 6) LN1 -> y1 (+bf16 mirror) ----
  layernorm_kernel<false, true><<<NTOK, 256, 0, stream>>>(y0, nullptr, g1, be1, y1, y1_bf);

  // ---- 7-10) top-1 routing ----
  moe_init<<<1, NEXP, 0, stream>>>(counts);
  gate_kernel<<<NTOK / 128, 128, 0, stream>>>(y1, Wg, expert_id, counts);
  moe_prefix<<<1, 1, 0, stream>>>(counts, ebase, cursor);
  moe_scatter<<<NTOK / 128, 128, 0, stream>>>(expert_id, cursor, perm);

  // ---- 11) h = relu(gather(y1) @ W1[e] + b1[e]) (bf16 out) ----
  moe_gemm<true, false, true, true><<<dim3(DFF / 64, NTOK / 128, NEXP), 256, 0, stream>>>(
      y1_bf, W1T, b1, h_bf, perm, ebase,
      DFF, D_MODEL, D_MODEL, DFF, (long long)DFF * D_MODEL);

  // ---- 12) moe[token] = h @ W2[e] + b2[e] (scatter, fp32 out) ----
  moe_gemm<false, true, false, false><<<dim3(D_MODEL / 64, NTOK / 128, NEXP), 256, 0, stream>>>(
      h_bf, W2T, b2, moe, perm, ebase,
      D_MODEL, DFF, DFF, D_MODEL, (long long)D_MODEL * DFF);

  // ---- 13) LN2(y1 + moe) -> out ----
  layernorm_kernel<true, false><<<NTOK, 256, 0, stream>>>(moe, y1, g2, be2, out, nullptr);
}
